// IFL_8486855377444
// MI455X (gfx1250) — matrix operand order repaired and checked by reading, not compiled, not run on hardware
//
#include <hip/hip_runtime.h>
#include <hip/hip_bf16.h>

#define BN 4096
#define FUC 10
#define FIC 30
#define HID 256
#define NENV 4
#define TEMPR 0.2f
#define ALPHA_C 0.1f
#define BETA_C 0.1f
#define SIGMA_C 0.5f
#define KMIT 10

typedef __attribute__((ext_vector_type(16))) __bf16 v16bf;
typedef __attribute__((ext_vector_type(8)))  __bf16 v8bf;
typedef __attribute__((ext_vector_type(8)))  float  v8f;
typedef __bf16 bf16;

union frag16 { v16bf v; v8bf h[2]; };

// A fragment (16x32 bf16, ISA 7.12.2): per lane, elements 0..7 hold K=kk+half*8+0..7,
// elements 8..15 hold K=kk+16+half*8+0..7. p must point at (row base + kk + half*8).
__device__ __forceinline__ v16bf ld_frag_a(const bf16* p) {
    frag16 f;
    f.h[0] = *(const v8bf*)(p);
    f.h[1] = *(const v8bf*)(p + 16);
    return f.v;
}
// B fragment (32x16 bf16): per lane, elements 0..7 hold K=kk+half*8+0..7,
// elements 8..15 hold K=kk+16+half*8+0..7. p must point at (col base + kk + half*8).
__device__ __forceinline__ v16bf ld_frag_b(const bf16* p) {
    frag16 f;
    f.h[0] = *(const v8bf*)(p);
    f.h[1] = *(const v8bf*)(p + 16);
    return f.v;
}

// ---------------------------------------------------------------------------
// bf16 GEMM, f32 accumulate. A row-major [M,K] (lda), Bt = B^T row-major [N,K]
// (ldb). One wave computes a 32x64 C block: 2 M-tiles x 4 N-tiles = 8 WMMA
// accumulators; each B fragment is reused by both M-tiles.
// grid = (M/32, N/64), block = 32. All lda/ldb/K multiples of 8/32.
// C = alpha*A*B + bias[n]; optional f32 (Cf,ldc), bf16 (Cb,ldc), bf16
// transposed (CbT, ldcT: CbT[n*ldcT+m]).
// ---------------------------------------------------------------------------
__global__ void gemm_wmma_bf16(const bf16* __restrict__ A, size_t lda,
                               const bf16* __restrict__ Bt, size_t ldb,
                               float* __restrict__ Cf, bf16* __restrict__ Cb,
                               bf16* __restrict__ CbT,
                               const float* __restrict__ bias, float alpha,
                               int K, int ldc, int ldcT)
{
    int lane = threadIdx.x;
    int half = lane >> 4;
    int lrow = lane & 15;
    size_t m0 = (size_t)blockIdx.x * 32 + lrow;
    size_t n0 = (size_t)blockIdx.y * 64;
    v8f acc[2][4] = {};
    const bf16* arow0 = A + m0 * lda + half * 8;
    const bf16* arow1 = arow0 + (size_t)16 * lda;
    for (int kk = 0; kk < K; kk += 32) {
        v16bf a0 = ld_frag_a(arow0 + kk);
        v16bf a1 = ld_frag_a(arow1 + kk);
        // pull the next A cachelines toward the WGP while WMMAs execute
        __builtin_prefetch((const void*)(arow0 + kk + 128), 0, 1);
        __builtin_prefetch((const void*)(arow1 + kk + 128), 0, 1);
#pragma unroll
        for (int jt = 0; jt < 4; ++jt) {
            size_t n = n0 + jt * 16 + lrow;
            v16bf b = ld_frag_b(Bt + n * ldb + kk + half * 8);
            acc[0][jt] = __builtin_amdgcn_wmma_f32_16x16x32_bf16(false, a0, false, b,
                                                                 (short)0, acc[0][jt], false, false);
            acc[1][jt] = __builtin_amdgcn_wmma_f32_16x16x32_bf16(false, a1, false, b,
                                                                 (short)0, acc[1][jt], false, false);
        }
    }
#pragma unroll
    for (int mt = 0; mt < 2; ++mt) {
#pragma unroll
        for (int jt = 0; jt < 4; ++jt) {
#pragma unroll
            for (int v = 0; v < 8; ++v) {
                size_t row = (size_t)blockIdx.x * 32 + mt * 16 + v + half * 8;
                size_t col = n0 + jt * 16 + lrow;
                float val = acc[mt][jt][v] * alpha + (bias ? bias[col] : 0.0f);
                if (Cf)  Cf[row * ldc + col] = val;
                if (Cb)  Cb[row * ldc + col] = (bf16)val;
                if (CbT) CbT[col * ldcT + row] = (bf16)val;
            }
        }
    }
}

// ---------------------------------------------------------------------------
// Fused mask-gradient: q = h[B,64] @ Weff^T tile, immediately dotted with E.
// gm[f] += sum_{c,k} E[c, f*64+k] * (h[c,:] . Weff[f*64+k,:])
// grid = (B/16, F), one wave per block, 8 WMMAs per wave.
// ---------------------------------------------------------------------------
__global__ void grad_fused(const bf16* __restrict__ hb,     // [B,64]
                           const bf16* __restrict__ Weffb,  // [F*64, 64] row-major
                           const bf16* __restrict__ E,      // [B, F*64]
                           int F, float* __restrict__ gm)   // [F]
{
    int lane = threadIdx.x;
    int half = lane >> 4;
    int lrow = lane & 15;
    int tm = blockIdx.x;
    int f  = blockIdx.y;
    size_t m = (size_t)tm * 16 + lrow;
    v8f acc[4] = {};
    const bf16* arow = hb + m * 64 + half * 8;
#pragma unroll
    for (int kk = 0; kk < 64; kk += 32) {
        v16bf a = ld_frag_a(arow + kk);
#pragma unroll
        for (int jt = 0; jt < 4; ++jt) {
            size_t n = (size_t)f * 64 + jt * 16 + lrow;  // row of Weff == col of q
            v16bf b = ld_frag_b(Weffb + n * 64 + kk + half * 8);
            acc[jt] = __builtin_amdgcn_wmma_f32_16x16x32_bf16(false, a, false, b,
                                                              (short)0, acc[jt], false, false);
        }
    }
    float part = 0.0f;
#pragma unroll
    for (int jt = 0; jt < 4; ++jt) {
#pragma unroll
        for (int v = 0; v < 8; ++v) {
            size_t row = (size_t)tm * 16 + v + half * 8;
            size_t col = (size_t)f * 64 + jt * 16 + lrow;
            part += acc[jt][v] * (float)E[row * (size_t)(F * 64) + col];
        }
    }
    __shared__ float red[32];
    red[lane] = part;
    __syncthreads();
    if (lane == 0) {
        float s = 0.0f;
        for (int i = 0; i < 32; ++i) s += red[i];
        atomicAdd(&gm[f], s);
    }
}

// ---------------------------------------------------------------------------
// Elementwise / small kernels
// ---------------------------------------------------------------------------
__global__ void zerof(float* p, size_t n) {
    size_t i = (size_t)blockIdx.x * blockDim.x + threadIdx.x;
    if (i < n) p[i] = 0.0f;
}

__global__ void f2bf(const float* __restrict__ in, bf16* __restrict__ out, size_t n) {
    size_t i = (size_t)blockIdx.x * blockDim.x + threadIdx.x;
    if (i < n) out[i] = (bf16)in[i];
}

// transpose-convert: in f32 [R,Cc] -> out bf16 [Cc,R]
__global__ void f2bfT(const float* __restrict__ in, bf16* __restrict__ out, int R, int Cc) {
    size_t i = (size_t)blockIdx.x * blockDim.x + threadIdx.x;
    if (i >= (size_t)R * Cc) return;
    int c = (int)(i % Cc);
    int r = (int)(i / Cc);
    out[(size_t)c * R + r] = (bf16)in[i];
}

__global__ void build_masks(const float* mu, const float* mi, const float* noise,
                            float* fmu, float* cmu, float* fmi, float* cmi, float* pv)
{
    if (threadIdx.x == 0 && blockIdx.x == 0) {
        for (int f = 0; f < FUC; ++f) {
            float pre = mu[f] + SIGMA_C * noise[f];
            fmu[f] = fminf(fmaxf(pre, 0.0f), 1.0f);
            cmu[f] = (pre > 0.0f && pre < 1.0f) ? 1.0f : 0.0f;
        }
        float mean = 0.0f;
        for (int f = 0; f < FIC; ++f) mean += mi[f];
        mean /= (float)FIC;
        for (int f = 0; f < FIC; ++f) {
            fmi[f] = fminf(fmaxf(mi[f], 0.0f), 1.0f);
            cmi[f] = (mi[f] > 0.0f && mi[f] < 1.0f) ? 1.0f : 0.0f;
            pv[f]  = (mi[f] > mean) ? 1.0f : 0.0f;
        }
    }
}

// Vectorized gather: each thread produces 8 bf16 (two float4 emb loads).
// out = emb[feats[b,f]][k] * vals[b,f] * fm[f] * extra[f]; out_raw = emb*val.
__global__ void gather_scale(const int* __restrict__ feats, const float* __restrict__ vals,
                             const float* __restrict__ emb,
                             const float* __restrict__ fm, const float* __restrict__ extra,
                             bf16* __restrict__ out, bf16* __restrict__ out_raw, int F)
{
    size_t i8 = (size_t)blockIdx.x * blockDim.x + threadIdx.x;
    size_t total8 = (size_t)BN * F * 8;
    if (i8 >= total8) return;
    int k8 = (int)(i8 & 7);            // which 8-wide chunk of the 64 dims
    size_t bf_ = i8 >> 3;
    int f = (int)(bf_ % F);
    size_t b = bf_ / F;
    int row = feats[b * F + f];
    const float4* ep = (const float4*)(emb + (size_t)row * 64 + k8 * 8);
    float4 e0 = ep[0], e1 = ep[1];
    float v  = vals[b * F + f];
    float sc = v * (fm ? fm[f] : 1.0f) * (extra ? extra[f] : 1.0f);
    float ev[8] = {e0.x, e0.y, e0.z, e0.w, e1.x, e1.y, e1.z, e1.w};
    size_t o = bf_ * 64 + (size_t)k8 * 8;
    if (out_raw) {
        v8bf rw;
#pragma unroll
        for (int j = 0; j < 8; ++j) rw[j] = (bf16)(ev[j] * v);
        *(v8bf*)(out_raw + o) = rw;
    }
    v8bf ow;
#pragma unroll
    for (int j = 0; j < 8; ++j) ow[j] = (bf16)(ev[j] * sc);
    *(v8bf*)(out + o) = ow;
}

// beff[n] = b2[n] + sum_k b1[k]*W2[k,n]   (block of 64 threads)
__global__ void beff_k(const float* __restrict__ b1, const float* __restrict__ W2,
                       const float* __restrict__ b2, float* __restrict__ out)
{
    int n = threadIdx.x;
    float s = b2[n];
    for (int k = 0; k < HID; ++k) s += b1[k] * W2[(size_t)k * 64 + n];
    out[n] = s;
}

// row L2 normalize (64-wide rows); writes f32, bf16, optional bf16-transposed, norm
__global__ void rownorm(const float* __restrict__ X, float* __restrict__ outf,
                        bf16* __restrict__ outb, bf16* __restrict__ outbT,
                        float* __restrict__ norms)
{
    int b = blockIdx.x, d = threadIdx.x;
    __shared__ float red[64];
    size_t i = (size_t)b * 64 + d;
    float v = X[i];
    red[d] = v * v;
    __syncthreads();
    for (int s = 32; s > 0; s >>= 1) { if (d < s) red[d] += red[d + s]; __syncthreads(); }
    float nrm = fmaxf(sqrtf(red[0]), 1e-12f);
    float nv = v / nrm;
    if (outf) outf[i] = nv;
    outb[i] = (bf16)nv;
    if (outbT) outbT[(size_t)d * BN + b] = (bf16)nv;
    if (norms && d == 0) norms[b] = nrm;
}

// Per-row softmax/LSE over bf16 logits S[BN,BN] (vectorized row scans).
// Optionally overwrites row with P (bf16) and scatters P^T.
// Accumulates sum_b (lse_b - pos_b) into acc.
__global__ void ssl_row(bf16* __restrict__ S, bf16* __restrict__ PT,
                        const float* __restrict__ n1, const float* __restrict__ n2,
                        float* __restrict__ acc, int writeP)
{
    int b = blockIdx.x, tid = threadIdx.x;  // 256 threads
    __shared__ float red[256];
    size_t base = (size_t)b * BN;
    const v8bf* Sv = (const v8bf*)(S + base);
    float mx = -3.0e38f;
    for (int c8 = tid; c8 < BN / 8; c8 += 256) {
        v8bf w = Sv[c8];
#pragma unroll
        for (int j = 0; j < 8; ++j) mx = fmaxf(mx, (float)w[j]);
    }
    red[tid] = mx; __syncthreads();
    for (int s = 128; s > 0; s >>= 1) { if (tid < s) red[tid] = fmaxf(red[tid], red[tid + s]); __syncthreads(); }
    mx = red[0]; __syncthreads();
    float sum = 0.0f;
    for (int c8 = tid; c8 < BN / 8; c8 += 256) {
        v8bf w = Sv[c8];
#pragma unroll
        for (int j = 0; j < 8; ++j) sum += expf((float)w[j] - mx);
    }
    red[tid] = sum; __syncthreads();
    for (int s = 128; s > 0; s >>= 1) { if (tid < s) red[tid] += red[tid + s]; __syncthreads(); }
    float tot = red[0]; __syncthreads();
    if (writeP) {
        float inv = 1.0f / tot;
        for (int c8 = tid; c8 < BN / 8; c8 += 256) {
            v8bf w = Sv[c8];
            v8bf pw;
#pragma unroll
            for (int j = 0; j < 8; ++j) {
                float pval = expf((float)w[j] - mx) * inv;
                pw[j] = (bf16)pval;
                PT[(size_t)(c8 * 8 + j) * BN + b] = (bf16)pval;
            }
            *(v8bf*)(S + base + (size_t)c8 * 8) = pw;
        }
    }
    float pp = (tid < 64) ? n1[(size_t)b * 64 + tid] * n2[(size_t)b * 64 + tid] : 0.0f;
    red[tid] = pp; __syncthreads();
    for (int s = 128; s > 0; s >>= 1) { if (tid < s) red[tid] += red[tid + s]; __syncthreads(); }
    if (tid == 0) {
        float pos = red[0] * (1.0f / TEMPR);
        float lse = mx + logf(tot);
        atomicAdd(acc, lse - pos);
    }
}

// ------------------------------ k-means -----------------------------------
__global__ void km_init(const float* __restrict__ ur, const float* __restrict__ ir,
                        float* __restrict__ C)
{
    int i = blockIdx.x * blockDim.x + threadIdx.x;
    if (i >= NENV * 128) return;
    int t = i / 128, d = i % 128;
    C[i] = (d < 64) ? ur[(size_t)t * 64 + d] : ir[(size_t)t * 64 + (d - 64)];
}

__global__ void km_step(const float* __restrict__ ur, const float* __restrict__ ir,
                        const float* __restrict__ C, float* __restrict__ sums,
                        float* __restrict__ cnts, int* __restrict__ ids, int write_ids)
{
    int b = blockIdx.x * blockDim.x + threadIdx.x;
    if (b >= BN) return;
    float best = 3.0e38f; int bi = 0;
    for (int t = 0; t < NENV; ++t) {
        float d2 = 0.0f;
        for (int d = 0; d < 64; ++d) { float x = ur[(size_t)b * 64 + d] - C[t * 128 + d];      d2 += x * x; }
        for (int d = 0; d < 64; ++d) { float x = ir[(size_t)b * 64 + d] - C[t * 128 + 64 + d]; d2 += x * x; }
        if (d2 < best) { best = d2; bi = t; }
    }
    if (write_ids) { ids[b] = bi; return; }
    for (int d = 0; d < 64; ++d) {
        atomicAdd(&sums[bi * 128 + d],      ur[(size_t)b * 64 + d]);
        atomicAdd(&sums[bi * 128 + 64 + d], ir[(size_t)b * 64 + d]);
    }
    atomicAdd(&cnts[bi], 1.0f);
}

__global__ void km_finalize(const float* __restrict__ sums, const float* __restrict__ cnts,
                            float* __restrict__ C)
{
    int i = blockIdx.x * blockDim.x + threadIdx.x;
    if (i >= NENV * 128) return;
    C[i] = sums[i] / fmaxf(cnts[i / 128], 1.0f);
}

__global__ void env_count(const int* __restrict__ ids, float* __restrict__ cnt) {
    int b = blockIdx.x * blockDim.x + threadIdx.x;
    if (b < BN) atomicAdd(&cnt[ids[b]], 1.0f);
}

__global__ void env_alpha(const int* __restrict__ ids, const float* __restrict__ cnt,
                          float* __restrict__ alphas) {
    int b = blockIdx.x * blockDim.x + threadIdx.x;
    if (b >= BN) return;
    for (int t = 0; t < NENV; ++t)
        alphas[(size_t)t * BN + b] = (ids[b] == t) ? 1.0f / fmaxf(cnt[t], 1.0f) : 0.0f;
}

// ----------------------- per-env gradient pieces ---------------------------
// M1^T[d, b] = alpha[b] * n1[b, d]
__global__ void build_m1T(const float* __restrict__ alpha, const float* __restrict__ n1,
                          bf16* __restrict__ m1t) {
    size_t i = (size_t)blockIdx.x * blockDim.x + threadIdx.x;
    if (i >= (size_t)BN * 64) return;
    size_t b = i >> 6;
    int d = (int)(i & 63);
    m1t[(size_t)d * BN + b] = (bf16)(alpha[b] * n1[i]);
}

// hu = ( g - (g.n1) n1 ) / ||ur||,  g = alpha * (r - n2) / temp
__global__ void build_hu(const float* __restrict__ r, const float* __restrict__ n1,
                         const float* __restrict__ n2, const float* __restrict__ normu,
                         const float* __restrict__ alpha, bf16* __restrict__ hub)
{
    int b = blockIdx.x, d = threadIdx.x;
    __shared__ float red[64];
    size_t i = (size_t)b * 64 + d;
    float g = alpha[b] * (r[i] - n2[i]) * (1.0f / TEMPR);
    red[d] = g * n1[i]; __syncthreads();
    for (int s = 32; s > 0; s >>= 1) { if (d < s) red[d] += red[d + s]; __syncthreads(); }
    float dot = red[0];
    hub[i] = (bf16)((g - dot * n1[i]) / normu[b]);
}

// hi = ( G - (G.n2) n2 ) / ||ir||,  G = (At - alpha * n1) / temp
__global__ void build_hi(const float* __restrict__ At, const float* __restrict__ n1,
                         const float* __restrict__ n2, const float* __restrict__ normi,
                         const float* __restrict__ alpha, bf16* __restrict__ hib)
{
    int b = blockIdx.x, d = threadIdx.x;
    __shared__ float red[64];
    size_t i = (size_t)b * 64 + d;
    float g = (At[i] - alpha[b] * n1[i]) * (1.0f / TEMPR);
    red[d] = g * n2[i]; __syncthreads();
    for (int s = 32; s > 0; s >>= 1) { if (d < s) red[d] += red[d + s]; __syncthreads(); }
    float dot = red[0];
    hib[i] = (bf16)((g - dot * n2[i]) / normi[b]);
}

__global__ void finalize(const float* __restrict__ gmu, const float* __restrict__ gmi,
                         const float* __restrict__ cmu, const float* __restrict__ cmi,
                         const float* __restrict__ main_sum, const float* __restrict__ csum,
                         float* __restrict__ out)
{
    if (threadIdx.x || blockIdx.x) return;
    float guavg[FUC], giavg[FIC];
    for (int f = 0; f < FUC; ++f) {
        float a = 0.0f;
        for (int t = 0; t < NENV; ++t) a += gmu[t * FUC + f] * cmu[f];
        guavg[f] = a / (float)NENV;
    }
    for (int f = 0; f < FIC; ++f) {
        float a = 0.0f;
        for (int t = 0; t < NENV; ++t) a += gmi[t * FIC + f] * cmi[f];
        giavg[f] = a / (float)NENV;
    }
    float pen = 0.0f;
    for (int t = 0; t < NENV; ++t) {
        for (int f = 0; f < FIC; ++f) { float d = gmi[t * FIC + f] * cmi[f] - giavg[f]; pen += d * d; }
        for (int f = 0; f < FUC; ++f) { float d = gmu[t * FUC + f] * cmu[f] - guavg[f]; pen += d * d; }
    }
    out[0] = main_sum[0] / (float)BN + ALPHA_C * (csum[0] / (float)BN) + BETA_C * pen;
}

// ---------------------------------------------------------------------------
extern "C" void kernel_launch(void* const* d_in, const int* in_sizes, int n_in,
                              void* d_out, int out_size, void* d_ws, size_t ws_size,
                              hipStream_t stream)
{
    (void)in_sizes; (void)n_in; (void)out_size; (void)ws_size;
    const int*   user_feats = (const int*)d_in[0];
    const float* user_vals  = (const float*)d_in[1];
    const int*   item_feats = (const int*)d_in[2];
    const float* item_vals  = (const float*)d_in[3];
    const int*   item_pos   = (const int*)d_in[4];
    const float* item_pvals = (const float*)d_in[5];
    const float* emb    = (const float*)d_in[8];
    const float* mask_u = (const float*)d_in[9];
    const float* mask_i = (const float*)d_in[10];
    const float* noise  = (const float*)d_in[11];
    const float* Wu1 = (const float*)d_in[12]; const float* bu1 = (const float*)d_in[13];
    const float* Wu2 = (const float*)d_in[14]; const float* bu2 = (const float*)d_in[15];
    const float* Wi1 = (const float*)d_in[16]; const float* bi1 = (const float*)d_in[17];
    const float* Wi2 = (const float*)d_in[18]; const float* bi2 = (const float*)d_in[19];
    float* out = (float*)d_out;

    const int KU = FUC * 64;   // 640
    const int KI = FIC * 64;   // 1920

    char* wp = (char*)d_ws;
    auto alloc = [&](size_t bytes) -> void* {
        void* q = (void*)wp;
        wp += (bytes + 255) & ~(size_t)255;
        return q;
    };

    bf16* Xu  = (bf16*)alloc((size_t)BN * KU * 2);
    bf16* Eu  = (bf16*)alloc((size_t)BN * KU * 2);
    bf16* Xi  = (bf16*)alloc((size_t)BN * KI * 2);
    bf16* Ei  = (bf16*)alloc((size_t)BN * KI * 2);
    bf16* Xa  = (bf16*)alloc((size_t)BN * KI * 2);
    bf16* Xf  = (bf16*)alloc((size_t)BN * KI * 2);
    bf16* S   = (bf16*)alloc((size_t)BN * BN * 2);   // logits -> P (in place); reused for S2
    bf16* PT  = (bf16*)alloc((size_t)BN * BN * 2);   // P^T for the per-env GEMMs
    bf16* Wu1b = (bf16*)alloc((size_t)KU * HID * 2);
    bf16* Wu2tb = (bf16*)alloc((size_t)HID * 64 * 2);   // [64, 256]
    bf16* Wi1b = (bf16*)alloc((size_t)KI * HID * 2);
    bf16* Wi2tb = (bf16*)alloc((size_t)HID * 64 * 2);   // [64, 256]
    bf16* WeffUb  = (bf16*)alloc((size_t)KU * 64 * 2);  // [640, 64]
    bf16* WeffUTb = (bf16*)alloc((size_t)KU * 64 * 2);  // [64, 640]
    bf16* WeffIb  = (bf16*)alloc((size_t)KI * 64 * 2);  // [1920, 64]
    bf16* WeffITb = (bf16*)alloc((size_t)KI * 64 * 2);  // [64, 1920]
    float* beffu = (float*)alloc(64 * 4);
    float* beffi = (float*)alloc(64 * 4);
    float* ur = (float*)alloc((size_t)BN * 64 * 4);
    float* ir = (float*)alloc((size_t)BN * 64 * 4);
    float* n1 = (float*)alloc((size_t)BN * 64 * 4);
    float* n2 = (float*)alloc((size_t)BN * 64 * 4);
    bf16* n1b  = (bf16*)alloc((size_t)BN * 64 * 2);
    bf16* n2b  = (bf16*)alloc((size_t)BN * 64 * 2);
    bf16* n2bT = (bf16*)alloc((size_t)BN * 64 * 2);  // [64, BN]
    float* normu = (float*)alloc(BN * 4);
    float* normi = (float*)alloc(BN * 4);
    float* rmat = (float*)alloc((size_t)BN * 64 * 4);   // r = P @ n2
    float* za = (float*)alloc((size_t)BN * 64 * 4);
    float* zf = (float*)alloc((size_t)BN * 64 * 4);
    float* zan = (float*)alloc((size_t)BN * 64 * 4);
    float* zfn = (float*)alloc((size_t)BN * 64 * 4);
    bf16* zanb = (bf16*)alloc((size_t)BN * 64 * 2);
    bf16* zfnb = (bf16*)alloc((size_t)BN * 64 * 2);
    float* At = (float*)alloc((size_t)BN * 64 * 4);
    bf16* M1tb = (bf16*)alloc((size_t)BN * 64 * 2);  // [64, BN]
    bf16* hub = (bf16*)alloc((size_t)BN * 64 * 2);
    bf16* hib = (bf16*)alloc((size_t)BN * 64 * 2);
    float* alphas = (float*)alloc((size_t)NENV * BN * 4);
    int*   ids = (int*)alloc(BN * 4);
    float* kms = (float*)alloc((NENV * 128 + NENV) * 4);  // sums + cnts
    float* Cc  = (float*)alloc(NENV * 128 * 4);           // centroids
    float* mb  = (float*)alloc(128 * 4);                  // masks block
    float* fmu = mb, *cmu = mb + 16, *fmi = mb + 32, *cmi = mb + 64, *pv = mb + 96;
    float* accums = (float*)alloc(256 * 4);
    float* main_sum = accums + 0;
    float* csum     = accums + 1;
    float* gmu      = accums + 2;    // [NENV*FUC] = 40
    float* gmi      = accums + 48;   // [NENV*FIC] = 120
    float* envcnt   = accums + 176;  // [NENV]

    // ---- 0: zero accumulators (ws not re-poisoned between replays) ----
    zerof<<<1, 256, 0, stream>>>(accums, 256);

    // ---- 1: masks / pos_vals ----
    build_masks<<<1, 32, 0, stream>>>(mask_u, mask_i, noise, fmu, cmu, fmi, cmi, pv);

    // ---- 2: weights -> bf16 (W2 transposed), Weff = W1@W2 (+transposed), beff ----
    {
        size_t n;
        n = (size_t)KU * HID; f2bf<<<(int)((n + 255) / 256), 256, 0, stream>>>(Wu1, Wu1b, n);
        n = (size_t)KI * HID; f2bf<<<(int)((n + 255) / 256), 256, 0, stream>>>(Wi1, Wi1b, n);
        n = (size_t)HID * 64;
        f2bfT<<<(int)((n + 255) / 256), 256, 0, stream>>>(Wu2, Wu2tb, HID, 64);
        f2bfT<<<(int)((n + 255) / 256), 256, 0, stream>>>(Wi2, Wi2tb, HID, 64);
    }
    gemm_wmma_bf16<<<dim3(KU / 32, 1), 32, 0, stream>>>(Wu1b, HID, Wu2tb, HID,
        nullptr, WeffUb, WeffUTb, nullptr, 1.0f, HID, 64, KU);
    gemm_wmma_bf16<<<dim3(KI / 32, 1), 32, 0, stream>>>(Wi1b, HID, Wi2tb, HID,
        nullptr, WeffIb, WeffITb, nullptr, 1.0f, HID, 64, KI);
    beff_k<<<1, 64, 0, stream>>>(bu1, Wu2, bu2, beffu);
    beff_k<<<1, 64, 0, stream>>>(bi1, Wi2, bi2, beffi);

    // ---- 3: gathers (masked X + raw E, anchor/full) ----
    {
        size_t tu = (size_t)BN * FUC * 8, ti = (size_t)BN * FIC * 8;
        gather_scale<<<(int)((tu + 255) / 256), 256, 0, stream>>>(user_feats, user_vals, emb, fmu, nullptr, Xu, Eu, FUC);
        gather_scale<<<(int)((ti + 255) / 256), 256, 0, stream>>>(item_feats, item_vals, emb, fmi, nullptr, Xi, Ei, FIC);
        gather_scale<<<(int)((ti + 255) / 256), 256, 0, stream>>>(item_pos, item_pvals, emb, nullptr, pv, Xf, Xa, FIC);
    }

    // ---- 4: representations ur/ir ----
    gemm_wmma_bf16<<<dim3(BN / 32, 1), 32, 0, stream>>>(Xu, KU, WeffUTb, KU,
        ur, nullptr, nullptr, beffu, 1.0f, KU, 64, 0);
    gemm_wmma_bf16<<<dim3(BN / 32, 1), 32, 0, stream>>>(Xi, KI, WeffITb, KI,
        ir, nullptr, nullptr, beffi, 1.0f, KI, 64, 0);

    // ---- 5: normalize ----
    rownorm<<<BN, 64, 0, stream>>>(ur, n1, n1b, nullptr, normu);
    rownorm<<<BN, 64, 0, stream>>>(ir, n2, n2b, n2bT, normi);

    // ---- 6: logits S = n1@n2^T / temp, softmax -> P (+P^T), main loss ----
    gemm_wmma_bf16<<<dim3(BN / 32, BN / 64), 32, 0, stream>>>(n1b, 64, n2b, 64,
        nullptr, S, nullptr, nullptr, 1.0f / TEMPR, 64, BN, 0);
    ssl_row<<<BN, 256, 0, stream>>>(S, PT, n1, n2, main_sum, 1);

    // ---- 7: r = P @ n2 (shared across envs) ----
    gemm_wmma_bf16<<<dim3(BN / 32, 1), 32, 0, stream>>>(S, BN, n2bT, BN,
        rmat, nullptr, nullptr, nullptr, 1.0f, BN, 64, 0);

    // ---- 8: k-means on concat(ur, ir) ----
    km_init<<<2, 256, 0, stream>>>(ur, ir, Cc);
    for (int it = 0; it < KMIT; ++it) {
        zerof<<<3, 256, 0, stream>>>(kms, NENV * 128 + NENV);
        km_step<<<BN / 256, 256, 0, stream>>>(ur, ir, Cc, kms, kms + NENV * 128, ids, 0);
        km_finalize<<<2, 256, 0, stream>>>(kms, kms + NENV * 128, Cc);
    }
    km_step<<<BN / 256, 256, 0, stream>>>(ur, ir, Cc, kms, kms + NENV * 128, ids, 1);
    env_count<<<BN / 256, 256, 0, stream>>>(ids, envcnt);
    env_alpha<<<BN / 256, 256, 0, stream>>>(ids, envcnt, alphas);

    // ---- 9: per-env mask gradients ----
    for (int t = 0; t < NENV; ++t) {
        const float* al = alphas + (size_t)t * BN;
        build_m1T<<<(BN * 64) / 256, 256, 0, stream>>>(al, n1, M1tb);
        // At[c,:] = sum_b alpha_b * P[b,c] * n1[b,:]  == P^T @ M1
        gemm_wmma_bf16<<<dim3(BN / 32, 1), 32, 0, stream>>>(PT, BN, M1tb, BN,
            At, nullptr, nullptr, nullptr, 1.0f, BN, 64, 0);
        build_hu<<<BN, 64, 0, stream>>>(rmat, n1, n2, normu, al, hub);
        build_hi<<<BN, 64, 0, stream>>>(At, n1, n2, normi, al, hib);
        grad_fused<<<dim3(BN / 16, FUC), 32, 0, stream>>>(hub, WeffUb, Eu, FUC, gmu + t * FUC);
        grad_fused<<<dim3(BN / 16, FIC), 32, 0, stream>>>(hib, WeffIb, Ei, FIC, gmi + t * FIC);
    }

    // ---- 10: contrastive SSL on items (reuses S buffer after grads done) ----
    gemm_wmma_bf16<<<dim3(BN / 32, 1), 32, 0, stream>>>(Xa, KI, WeffITb, KI,
        za, nullptr, nullptr, beffi, 1.0f, KI, 64, 0);
    gemm_wmma_bf16<<<dim3(BN / 32, 1), 32, 0, stream>>>(Xf, KI, WeffITb, KI,
        zf, nullptr, nullptr, beffi, 1.0f, KI, 64, 0);
    rownorm<<<BN, 64, 0, stream>>>(za, zan, zanb, nullptr, nullptr);
    rownorm<<<BN, 64, 0, stream>>>(zf, zfn, zfnb, nullptr, nullptr);
    gemm_wmma_bf16<<<dim3(BN / 32, BN / 64), 32, 0, stream>>>(zanb, 64, zfnb, 64,
        nullptr, S, nullptr, nullptr, 1.0f / TEMPR, 64, BN, 0);
    ssl_row<<<BN, 256, 0, stream>>>(S, PT, zan, zfn, csum, 0);

    // ---- 11: scalar output ----
    finalize<<<1, 1, 0, stream>>>(gmu, gmi, cmu, cmi, main_sum, csum, out);
}
